// BatchedFCN_43413529428144
// MI455X (gfx1250) — compile-verified, matrix operand order checked by reading, not run on hardware
//
#include <hip/hip_runtime.h>
#include <hip/hip_bf16.h>

// ---------------------------------------------------------------------------
// MI455X (gfx1250) batched-ensemble MLP.
//   - bf16 WMMA 16x16x32 (v_wmma_f32_16x16x32_bf16), f32 accumulate
//   - weights streamed from HBM exactly once via the Tensor Data Mover
//     (tensor_load_to_lds, TENSORcnt), double-buffered f32 stage in LDS,
//     vectorized LDS->LDS f32->bf16 repack
//   - one workgroup (8 wave32) per network; activations ping-pong in LDS
// ---------------------------------------------------------------------------

typedef __attribute__((ext_vector_type(16))) __bf16 v16bf;
typedef __attribute__((ext_vector_type(8)))  float  v8f;
typedef __attribute__((ext_vector_type(4)))  unsigned int v4u;
typedef __attribute__((ext_vector_type(8)))  int    v8i;
typedef __attribute__((ext_vector_type(4)))  int    v4i;

#define NNET  1024
#define BATCH 256

__device__ __forceinline__ unsigned short bf_rne(unsigned u) {
    // round-to-nearest-even f32 bits -> bf16 bits
    return (unsigned short)((u + 0x7FFFu + ((u >> 16) & 1u)) >> 16);
}

__device__ __forceinline__ __bf16 f2bf(float f) {
    unsigned short h = bf_rne(__builtin_bit_cast(unsigned, f));
    return __builtin_bit_cast(__bf16, h);
}

union FragBits { v16bf v; uint4 q[2]; };

// Load one 16x32 bf16 WMMA operand fragment for this lane.
// Per ISA layout: lane holds row (lane&15); VGPR0-3 = K[koff..koff+8),
// VGPR4-7 = K[koff+16..koff+24), koff = (lane>=16)*8.
__device__ __forceinline__ v16bf load_frag(const __bf16* p) {
    FragBits f;
    f.q[0] = *(const uint4*)(p);
    f.q[1] = *(const uint4*)(p + 16);
    return f.v;
}

// TDM: DMA one W chunk [OPAD rows][32 cols] f32 from global into LDS.
// D# per CDNA5 ISA:
//   group0: count=1 | lds_addr | global_addr[56:0] | type=2
//   group1: data_size=4B, tensor_dim0=rem_k (tail cols read as 0),
//           tensor_dim1=O (rows >= O read as 0), tile=32 x OPAD, stride0=K
__device__ __forceinline__ void tdm_load_w_chunk(
    const float* tile0, unsigned lds_off, int rem_k, int O, int OPAD, int K)
{
    unsigned long long ga = (unsigned long long)(uintptr_t)tile0;
    v4u g0;
    g0[0] = 1u;                                            // count=1, user D#
    g0[1] = lds_off;                                       // LDS byte address
    g0[2] = (unsigned)(ga & 0xFFFFFFFFu);
    g0[3] = (unsigned)((ga >> 32) & 0x01FFFFFFu) | (2u << 30);  // type=2
    v8i g1;
    g1[0] = (int)(2u << 16);                               // data_size=4B, mask=0
    g1[1] = (int)(((unsigned)rem_k & 0xFFFFu) << 16);      // tensor_dim0[15:0]
    g1[2] = (int)((((unsigned)rem_k >> 16) & 0xFFFFu) |
                  (((unsigned)O & 0xFFFFu) << 16));        // td0[31:16] | td1[15:0]
    g1[3] = (int)((((unsigned)O >> 16) & 0xFFFFu) |
                  (32u << 16));                            // td1[31:16] | tile_dim0=32
    g1[4] = (int)((unsigned)OPAD);                         // tile_dim1=OPAD, tile_dim2=0
    g1[5] = (int)(unsigned)K;                              // tensor_dim0_stride[31:0]
    g1[6] = 0;                                             // stride0 hi | stride1 lo
    g1[7] = 0;
    v4i z4 = {0, 0, 0, 0};
    v8i z8 = {0, 0, 0, 0, 0, 0, 0, 0};
    // 6-arg form (clang-23 / therock-10.0): (g0, g1, g2, g3, g4, cpol)
    __builtin_amdgcn_tensor_load_to_lds(g0, g1, z4, z4, z8, 0);
}

// One MLP layer: OUT[256][128] = act(A[256][*] @ W^T + b)
//  KCH : number of K chunks of 32;  OT : 16-wide output tiles (OPAD = OT*16)
//  TDM zero-fill + bias guard make all padded outputs exactly 0, so padded
//  columns are valid zero K-inputs for the next layer.
template <int KCH, int OT, bool RELU, bool LAST>
__device__ __forceinline__ void mlp_layer(
    const __bf16* __restrict__ A, int a_stride,
    const float*  __restrict__ W,     // global, [O][K] row-major f32
    const float*  __restrict__ bias,  // global, [O] f32
    int K, int O,
    __bf16* __restrict__ out,         // LDS, [256][128] bf16 (unused if LAST)
    float*  __restrict__ scores_out,  // global scores (only if LAST)
    __bf16* __restrict__ wbuf,        // LDS, [OPAD][32] bf16
    float*  __restrict__ sf32,        // LDS, [2][128*32] f32 TDM stage
    int net)
{
    const int tid  = threadIdx.x;
    const int lane = tid & 31;
    const int wv   = tid >> 5;           // 8 waves
    const int colp = lane & 15;
    const int koff = (lane >> 4) << 3;   // 0 or 8
    constexpr int OPAD = OT * 16;

    v8f acc[2][OT];
#pragma unroll
    for (int mt = 0; mt < 2; ++mt)
#pragma unroll
        for (int ot = 0; ot < OT; ++ot) acc[mt][ot] = (v8f)0.0f;

    // Prologue: kick off DMA of chunk 0 (wave 0 only; branch is wave-uniform).
    if (wv == 0) {
        tdm_load_w_chunk(W, (unsigned)(uintptr_t)&sf32[0], K, O, OPAD, K);
    }

    for (int kc = 0; kc < KCH; ++kc) {
        if (wv == 0) __builtin_amdgcn_s_wait_tensorcnt(0);
        __syncthreads();   // chunk kc landed in sf32[kc&1]; prev compute done

        // Prefetch next chunk into the other stage buffer while we work.
        if (wv == 0 && kc + 1 < KCH) {
            tdm_load_w_chunk(W + (kc + 1) * 32,
                             (unsigned)(uintptr_t)&sf32[((kc + 1) & 1) * (128 * 32)],
                             K - (kc + 1) * 32, O, OPAD, K);
        }

        // Vectorized LDS->LDS repack: 4 f32 -> 4 bf16 per iteration.
        {
            const unsigned* src = (const unsigned*)&sf32[(kc & 1) * (128 * 32)];
            uint2* dst = (uint2*)wbuf;
            for (int g = tid; g < OPAD * 8; g += 256) {
                uint4 w4 = *(const uint4*)(src + g * 4);
                uint2 p;
                p.x = (unsigned)bf_rne(w4.x) | ((unsigned)bf_rne(w4.y) << 16);
                p.y = (unsigned)bf_rne(w4.z) | ((unsigned)bf_rne(w4.w) << 16);
                dst[g] = p;
            }
        }
        __syncthreads();   // wbuf ready for all waves

        // A fragments for this wave's two M tiles
        v16bf a0 = load_frag(A + ((wv     ) * 16 + colp) * a_stride + kc * 32 + koff);
        v16bf a1 = load_frag(A + ((wv +  8) * 16 + colp) * a_stride + kc * 32 + koff);

#pragma unroll
        for (int ot = 0; ot < OT; ++ot) {
            v16bf b = load_frag(wbuf + (ot * 16 + colp) * 32 + koff);
            acc[0][ot] = __builtin_amdgcn_wmma_f32_16x16x32_bf16(
                false, a0, false, b, (short)0, acc[0][ot], false, false);
            acc[1][ot] = __builtin_amdgcn_wmma_f32_16x16x32_bf16(
                false, a1, false, b, (short)0, acc[1][ot], false, false);
        }
    }

    // Epilogue: C layout = lane (colp) is column, VGPR r is row (+8 for hi lanes)
    const int rbase = (lane >> 4) << 3;
#pragma unroll
    for (int mt = 0; mt < 2; ++mt) {
        int mrow = (wv + mt * 8) * 16 + rbase;
#pragma unroll
        for (int ot = 0; ot < OT; ++ot) {
            int col = ot * 16 + colp;
            float bv = (col < O) ? bias[col] : 0.0f;
#pragma unroll
            for (int r = 0; r < 8; ++r) {
                float v = acc[mt][ot][r] + bv;
                if (RELU) v = fmaxf(v, 0.0f);
                if (LAST) {
                    if (col == 0) scores_out[net * BATCH + mrow + r] = v;
                } else {
                    out[(mrow + r) * 128 + col] = f2bf(v);
                }
            }
        }
    }
}

__global__ __launch_bounds__(256) void mlp_main(
    const __bf16* __restrict__ xg,  // [256][512] bf16, zero-padded
    const float* __restrict__ w1, const float* __restrict__ b1,
    const float* __restrict__ w2, const float* __restrict__ b2,
    const float* __restrict__ w3, const float* __restrict__ b3,
    const float* __restrict__ w4, const float* __restrict__ b4,
    const float* __restrict__ w5, const float* __restrict__ b5,
    float* __restrict__ scores)
{
    __shared__ __align__(16) __bf16 actA[256 * 128];      // 64 KB
    __shared__ __align__(16) __bf16 actB[256 * 128];      // 64 KB
    __shared__ __align__(16) __bf16 wbuf[128 * 32];       //  8 KB
    __shared__ __align__(16) float  sf32[2 * 128 * 32];   // 32 KB TDM stage
    const int net = blockIdx.x;

    // L1: 500 -> 100  (K 512, O 128pad), A from global bf16 x
    mlp_layer<16, 8, true,  false>(xg,   512, w1 + net * 50000, b1 + net * 100, 500, 100, actA, nullptr, wbuf, sf32, net);
    // L2: 100 -> 100
    mlp_layer< 4, 8, true,  false>(actA, 128, w2 + net * 10000, b2 + net * 100, 100, 100, actB, nullptr, wbuf, sf32, net);
    // L3: 100 -> 100
    mlp_layer< 4, 8, true,  false>(actB, 128, w3 + net * 10000, b3 + net * 100, 100, 100, actA, nullptr, wbuf, sf32, net);
    // L4: 100 -> 50   (O 64pad)
    mlp_layer< 4, 4, true,  false>(actA, 128, w4 + net *  5000, b4 + net *  50, 100,  50, actB, nullptr, wbuf, sf32, net);
    // L5: 50 -> 1, no relu, write per-network scalars
    mlp_layer< 2, 1, false, true >(actB, 128, w5 + net *    50, b5 + net *   1,  50,   1, nullptr, scores, wbuf, sf32, net);
}

__global__ __launch_bounds__(256) void convert_x(
    const float* __restrict__ x, __bf16* __restrict__ xg)
{
    int row = blockIdx.x;
    for (int c = threadIdx.x; c < 512; c += 256)
        xg[row * 512 + c] = f2bf(c < 500 ? x[row * 500 + c] : 0.0f);
}

__global__ __launch_bounds__(256) void reduce_max(
    const float* __restrict__ scores, float* __restrict__ out)
{
    __shared__ float red[256];
    int b = blockIdx.x, t = threadIdx.x;
    float m = -3.402823466e38f;
    for (int n = t; n < NNET; n += 256) m = fmaxf(m, scores[n * BATCH + b]);
    red[t] = m;
    __syncthreads();
    for (int s = 128; s > 0; s >>= 1) {
        if (t < s) red[t] = fmaxf(red[t], red[t + s]);
        __syncthreads();
    }
    if (t == 0) out[b] = red[0];
}

extern "C" void kernel_launch(void* const* d_in, const int* in_sizes, int n_in,
                              void* d_out, int out_size, void* d_ws, size_t ws_size,
                              hipStream_t stream) {
    (void)in_sizes; (void)n_in; (void)out_size; (void)ws_size;
    const float* x  = (const float*)d_in[0];
    const float* w1 = (const float*)d_in[1];
    const float* b1 = (const float*)d_in[2];
    const float* w2 = (const float*)d_in[3];
    const float* b2 = (const float*)d_in[4];
    const float* w3 = (const float*)d_in[5];
    const float* b3 = (const float*)d_in[6];
    const float* w4 = (const float*)d_in[7];
    const float* b4 = (const float*)d_in[8];
    const float* w5 = (const float*)d_in[9];
    const float* b5 = (const float*)d_in[10];

    // workspace layout: [0, 256KB) padded bf16 x; [256KB, 256KB+1MB) scores
    __bf16* xg     = (__bf16*)d_ws;
    float*  scores = (float*)((char*)d_ws + (size_t)BATCH * 512 * sizeof(__bf16));

    convert_x<<<BATCH, 256, 0, stream>>>(x, xg);
    mlp_main <<<NNET,  256, 0, stream>>>(xg, w1, b1, w2, b2, w3, b3, w4, b4, w5, b5, scores);
    reduce_max<<<BATCH, 256, 0, stream>>>(scores, (float*)d_out);
}
